// MinGRUBase_72267119723113
// MI455X (gfx1250) — matrix operand order repaired and checked by reading, not compiled, not run on hardware
//
#include <hip/hip_runtime.h>
#include <hip/hip_bf16.h>
#include <math.h>

// ---------------------------------------------------------------------------
// minGRU, 4 layers, B=8 S=2048 D=H=1024.
// Per layer:
//   1) cvt inp f32->bf16, cvt W f32->bf16
//   2) WMMA GEMM (bf16 in, f32 acc): gh = inp @ W^T + b, fused epilogue
//      writes LC = -softplus(gate), LV = -softplus(-gate) + log_g(hidden)
//   3) chunked log-space scan (3 passes), exp + residual, finals
// Wave tile is 2Mx2N(gate)+2Mx2N(hidden) = 64 acc VGPRs (spill-free).
// All log-space math is branchless and uses the HW v_exp_f32/v_log_f32 path;
// softplus(+g) and softplus(-g) share one exp/log via lse = log(1+exp(-|g|)).
// ---------------------------------------------------------------------------

typedef __attribute__((ext_vector_type(16))) __bf16 v16bf;
typedef __attribute__((ext_vector_type(8)))  __bf16 v8bf;
typedef __attribute__((ext_vector_type(4)))  __bf16 v4bf;
typedef __attribute__((ext_vector_type(8)))  float  v8f;

#define BB 8
#define SS 2048
#define HH 1024
#define KK 1024
#define LLAYERS 4
#define MM (BB*SS)            /* 16384 rows of the GEMM */
#define NCHUNK 16
#define CLEN (SS/NCHUNK)      /* 128 */
#define NCHAIN (BB*HH)        /* 8192 scan chains */

// ---------------- math helpers (branchless, fast HW path) ------------------
// lse(x) = log(1 + exp(-|x|))  (shared kernel of softplus(+-x))
__device__ __forceinline__ float lse_absneg_f(float x) {
    return __logf(1.0f + __expf(-fabsf(x)));
}
__device__ __forceinline__ float softplus_f(float x) {
    return fmaxf(x, 0.0f) + lse_absneg_f(x);
}
// log g(x); g(x) = x + 0.5 for x>=0 else sigmoid(x). Branchless select:
// x<0 branch: -softplus(-x) = x - lse(x). Dead log(x+0.5) NaN selected away.
__device__ __forceinline__ float log_g_f(float x) {
    float neg = x - lse_absneg_f(x);
    float pos = __logf(x + 0.5f);
    return (x >= 0.0f) ? pos : neg;
}
__device__ __forceinline__ float logaddexp_f(float a, float b) {
    float mx = fmaxf(a, b);
    float mn = fminf(a, b);
    float r = mx + __logf(1.0f + __expf(mn - mx));
    return (mx == -__builtin_inff()) ? -__builtin_inff() : r;
}

// ---------------- f32 -> bf16 conversion (vector x4) -----------------------
__global__ __launch_bounds__(256)
void cvt_f32_bf16(const float* __restrict__ in, __bf16* __restrict__ out, int n4) {
    int i = blockIdx.x * blockDim.x + threadIdx.x;
    if (i >= n4) return;
    float4 v = ((const float4*)in)[i];
    v4bf o;
    o[0] = (__bf16)v.x; o[1] = (__bf16)v.y; o[2] = (__bf16)v.z; o[3] = (__bf16)v.w;
    ((v4bf*)out)[i] = o;
}

// ---------------- WMMA GEMM + fused minGRU epilogue ------------------------
// C[m, n] = sum_k A[m,k] * W[n,k] + bias[n], n in [0,2048)
// Workgroup: 8 waves arranged 4(M) x 2(N) -> 128 rows x 64 gate cols
// (+ paired 64 hidden cols, n+1024). Wave: 32x32 gate + 32x32 hidden.
// A/B fragments loaded straight from row-major bf16 global memory using the
// CDNA5 16-bit WMMA lane layouts (no LDS transpose needed).
__global__ __launch_bounds__(256)
void mingru_gemm(const __bf16* __restrict__ A,    // (MM, KK) bf16
                 const __bf16* __restrict__ Wb,   // (2048, KK) bf16
                 const float*  __restrict__ bias, // (2048)
                 float* __restrict__ LC,          // (MM, HH)
                 float* __restrict__ LV)          // (MM, HH)
{
    const int lane = threadIdx.x & 31;
    const int wave = threadIdx.x >> 5;
    const int half = lane >> 4;     // 0: lanes 0-15, 1: lanes 16-31
    const int l16  = lane & 15;
    const int m0 = blockIdx.x * 128 + (wave >> 1) * 32;  // 2 x 16-row tiles
    const int nG = blockIdx.y * 64 + (wave & 1) * 32;    // 2 x 16-col tiles

    v8f zf = {};
    v8f accG[2][2], accH[2][2];
#pragma unroll
    for (int i = 0; i < 2; ++i)
#pragma unroll
        for (int j = 0; j < 2; ++j) { accG[i][j] = zf; accH[i][j] = zf; }

    // A 16x32 bf16 fragment: lane m (=l16) holds K 0..7 (half=0) / 8..15
    // (half=1) in v0-3 and K 16..23 / 24..31 in v4-7  -> two b128 per frag.
    const __bf16* Abase = A + (size_t)(m0 + l16) * KK + half * 8;
    // B 32x16 bf16 fragment: lane n (=l16) holds K 0..7 (half=0) / 8..15
    // (half=1) in v0-3 and K 16..23 / 24..31 in v4-7 -> two b128 per frag.
    const __bf16* Bg0 = Wb + (size_t)(nG + l16) * KK + half * 8;
    const __bf16* Bg1 = Bg0 + (size_t)16 * KK;
    const __bf16* Bh0 = Bg0 + (size_t)HH * KK;   // hidden cols = gate cols + 1024
    const __bf16* Bh1 = Bg1 + (size_t)HH * KK;

    for (int k0 = 0; k0 < KK; k0 += 32) {
        v16bf afr[2];
#pragma unroll
        for (int i = 0; i < 2; ++i) {
            const __bf16* p = Abase + (size_t)i * 16 * KK + k0;
            v8bf lo = *(const v8bf*)(p);
            v8bf hi = *(const v8bf*)(p + 16);
#pragma unroll
            for (int e = 0; e < 8; ++e) { afr[i][e] = lo[e]; afr[i][8 + e] = hi[e]; }
        }
        v16bf bg[2], bh[2];
        {
            const __bf16* pg[2] = { Bg0 + k0, Bg1 + k0 };
            const __bf16* ph[2] = { Bh0 + k0, Bh1 + k0 };
#pragma unroll
            for (int j = 0; j < 2; ++j) {
                v8bf g0 = *(const v8bf*)(pg[j]);
                v8bf g1 = *(const v8bf*)(pg[j] + 16);
                v8bf h0 = *(const v8bf*)(ph[j]);
                v8bf h1 = *(const v8bf*)(ph[j] + 16);
#pragma unroll
                for (int e = 0; e < 8; ++e) {
                    bg[j][e] = g0[e]; bg[j][8 + e] = g1[e];
                    bh[j][e] = h0[e]; bh[j][8 + e] = h1[e];
                }
            }
        }
#pragma unroll
        for (int i = 0; i < 2; ++i)
#pragma unroll
            for (int j = 0; j < 2; ++j) {
                accG[i][j] = __builtin_amdgcn_wmma_f32_16x16x32_bf16(
                    false, afr[i], false, bg[j], (short)0, accG[i][j], false, false);
                accH[i][j] = __builtin_amdgcn_wmma_f32_16x16x32_bf16(
                    false, afr[i], false, bh[j], (short)0, accH[i][j], false, false);
            }
    }

    // Epilogue: C/D layout -> VGPR r holds row (r + 8*half), col = l16.
    // softplus(+g) and softplus(-g) share lse = log(1+exp(-|g|)).
#pragma unroll
    for (int j = 0; j < 2; ++j) {
        const int colG = nG + j * 16 + l16;
        const float bG = bias[colG];
        const float bH = bias[colG + HH];
#pragma unroll
        for (int i = 0; i < 2; ++i) {
#pragma unroll
            for (int r = 0; r < 8; ++r) {
                const int mrow = m0 + i * 16 + r + half * 8;
                const float g   = accG[i][j][r] + bG;
                const float hv  = accH[i][j][r] + bH;
                const float lse = lse_absneg_f(g);
                const size_t off = (size_t)mrow * HH + colG;
                LC[off] = -(fmaxf(g, 0.0f) + lse);                    // -softplus(g)
                LV[off] = -(fmaxf(-g, 0.0f) + lse) + log_g_f(hv);     // -softplus(-g)+log_g(h)
            }
        }
    }
}

// ---------------- chunked log-space scan ------------------------------------
// Recurrence: L_t = logaddexp(lc_t + L_{t-1}, lv_t), L_0 = log_g(h_prev)
// Pass 1: per (chain, chunk) compute (sumA = sum lc, q = chunk run from -inf)
__global__ __launch_bounds__(256)
void scan_pass1(const float* __restrict__ LC, const float* __restrict__ LV,
                float* __restrict__ CA, float* __restrict__ CQ)
{
    int idx = blockIdx.x * blockDim.x + threadIdx.x;   // 0 .. NCHAIN*NCHUNK-1
    int h = idx & (HH - 1);
    int t2 = idx >> 10;
    int chunk = t2 & (NCHUNK - 1);
    int b = t2 >> 4;
    int chain = b * HH + h;
    size_t base = ((size_t)b * SS + (size_t)chunk * CLEN) * HH + h;
    float sa = 0.0f, q = -__builtin_inff();
#pragma unroll 4
    for (int t = 0; t < CLEN; ++t) {
        float lc = LC[base], lv = LV[base];
        sa += lc;
        q = logaddexp_f(q + lc, lv);
        base += HH;
    }
    CA[(size_t)chunk * NCHAIN + chain] = sa;
    CQ[(size_t)chunk * NCHAIN + chain] = q;
}

// Pass 2: per chain, 16-step cross-chunk recurrence. Emits chunk start states
// and finals = exp(L_S).
__global__ __launch_bounds__(256)
void scan_pass2(const float* __restrict__ CA, const float* __restrict__ CQ,
                const float* __restrict__ hprev_l,  // (NCHAIN)
                float* __restrict__ CL, float* __restrict__ fin)
{
    int chain = blockIdx.x * blockDim.x + threadIdx.x;  // 0..NCHAIN-1
    float L = log_g_f(hprev_l[chain]);
#pragma unroll
    for (int c = 0; c < NCHUNK; ++c) {
        size_t o = (size_t)c * NCHAIN + chain;
        CL[o] = L;
        L = logaddexp_f(CA[o] + L, CQ[o]);
    }
    fin[chain] = __expf(L);
}

// Pass 3: replay each chunk from its start state, exp + residual add.
__global__ __launch_bounds__(256)
void scan_pass3(const float* __restrict__ LC, const float* __restrict__ LV,
                const float* __restrict__ CL,
                const float* __restrict__ inp, float* __restrict__ outp)
{
    int idx = blockIdx.x * blockDim.x + threadIdx.x;
    int h = idx & (HH - 1);
    int t2 = idx >> 10;
    int chunk = t2 & (NCHUNK - 1);
    int b = t2 >> 4;
    int chain = b * HH + h;
    float L = CL[(size_t)chunk * NCHAIN + chain];
    size_t base = ((size_t)b * SS + (size_t)chunk * CLEN) * HH + h;
#pragma unroll 4
    for (int t = 0; t < CLEN; ++t) {
        float lc = LC[base], lv = LV[base];
        L = logaddexp_f(lc + L, lv);
        outp[base] = __expf(L) + inp[base];
        base += HH;
    }
}

// ---------------------------------------------------------------------------
extern "C" void kernel_launch(void* const* d_in, const int* in_sizes, int n_in,
                              void* d_out, int out_size, void* d_ws, size_t ws_size,
                              hipStream_t stream)
{
    const float* x     = (const float*)d_in[0];  // (B,S,D)
    const float* hprev = (const float*)d_in[1];  // (L,B,1,H)
    const float* W0    = (const float*)d_in[2];  // (2H,D)
    const float* b0    = (const float*)d_in[3];  // (2H)
    const float* Wl    = (const float*)d_in[4];  // (L-1,2H,H)
    const float* bl    = (const float*)d_in[5];  // (L-1,2H)
    float* out = (float*)d_out;                              // (B,S,H)
    float* finals = out + (size_t)MM * HH;                   // (L,B,1,H)

    // workspace carve-up
    char* w = (char*)d_ws;
    size_t off = 0;
    __bf16* INPB = (__bf16*)(w + off); off += (size_t)MM * KK * sizeof(__bf16);      // 32 MiB
    __bf16* WB   = (__bf16*)(w + off); off += (size_t)2 * HH * KK * sizeof(__bf16);  // 4 MiB
    float* LC    = (float*)(w + off);  off += (size_t)MM * HH * sizeof(float);       // 64 MiB
    float* LV    = (float*)(w + off);  off += (size_t)MM * HH * sizeof(float);       // 64 MiB
    float* OUTB  = (float*)(w + off);  off += (size_t)MM * HH * sizeof(float);       // 64 MiB
    float* CA    = (float*)(w + off);  off += (size_t)NCHAIN * NCHUNK * sizeof(float);
    float* CQ    = (float*)(w + off);  off += (size_t)NCHAIN * NCHUNK * sizeof(float);
    float* CL    = (float*)(w + off);  off += (size_t)NCHAIN * NCHUNK * sizeof(float);

    const int nInp4 = MM * KK / 4;            // 4,194,304
    const int nW4   = 2 * HH * KK / 4;        // 524,288
    const dim3 gemmGrid(MM / 128, HH / 64);   // (128, 16)
    const int scanBlocks = (NCHAIN * NCHUNK) / 256;  // 512

    for (int l = 0; l < LLAYERS; ++l) {
        const float* inp  = (l == 0) ? x : OUTB;
        const float* W    = (l == 0) ? W0 : Wl + (size_t)(l - 1) * 2 * HH * KK;
        const float* bias = (l == 0) ? b0 : bl + (size_t)(l - 1) * 2 * HH;
        float* outp = (l == LLAYERS - 1) ? out : OUTB;

        cvt_f32_bf16<<<nInp4 / 256, 256, 0, stream>>>(inp, INPB, nInp4);
        cvt_f32_bf16<<<nW4 / 256, 256, 0, stream>>>(W, WB, nW4);
        mingru_gemm<<<gemmGrid, 256, 0, stream>>>(INPB, WB, bias, LC, LV);
        scan_pass1<<<scanBlocks, 256, 0, stream>>>(LC, LV, CA, CQ);
        scan_pass2<<<NCHAIN / 256, 256, 0, stream>>>(CA, CQ,
                                                     hprev + (size_t)l * NCHAIN,
                                                     CL, finals + (size_t)l * NCHAIN);
        scan_pass3<<<scanBlocks, 256, 0, stream>>>(LC, LV, CL, inp, outp);
    }
    (void)in_sizes; (void)n_in; (void)out_size; (void)ws_size;
}
